// PointerNet_231928234598
// MI455X (gfx1250) — matrix operand order repaired and checked by reading, not compiled, not run on hardware
//
#include <hip/hip_runtime.h>
#include <hip/hip_bf16.h>
#include <math.h>

// Problem constants (from reference)
#define LK 128
#define LV 128
#define NB 16      // N (batch)
#define DK 256     // KEY_SIZE == VALUE_SIZE == HIDDEN
#define HH 256

typedef __attribute__((ext_vector_type(16))) __bf16 v16bf;
typedef __attribute__((ext_vector_type(8)))  float  v8f;

#if __has_builtin(__builtin_amdgcn_tanhf)
  #define FAST_TANH(x) __builtin_amdgcn_tanhf(x)
#elif __has_builtin(__builtin_amdgcn_tanh_f32)
  #define FAST_TANH(x) __builtin_amdgcn_tanh_f32(x)
#else
// gfx1250 has a 32-bit transcendental TANH (ISA flow/sync §8.4 lists TANH
// among TRANS32 ops). Single-instruction fallback if no clang builtin.
static __device__ __forceinline__ float fast_tanh_dev(float x) {
    float r;
    asm("v_tanh_f32 %0, %1" : "=v"(r) : "v"(x));
    return r;
}
  #define FAST_TANH(x) fast_tanh_dev(x)
#endif

// ---------------------------------------------------------------------------
// GEMM: Y[m,h] = sum_d X[m,d] * W[h,d] + bias[h]
// X: (M=2048, K=256) row-major (key/value flattened over (L,N))
// W: (H=256, K=256) row-major  -> B[k,h] = W[h,k] (we index W by column h)
// One wave (32 lanes) computes one 16x16 tile of Y via 8x v_wmma_f32_16x16x32_bf16
// (K-loop fully unrolled -> compiler pipelines loads/cvt against the matrix pipe).
// Fragment packing per CDNA5 ISA 7.12.2 (wave32):
//   A (16x32 bf16): lane L -> M=L&15; half hg=L>>4; VGPR j<4: K={2j,2j+1}+8*hg,
//                   VGPR j>=4: K={16+2(j-4),...}+8*hg   (pairs contiguous in K)
//   B (32x16 bf16): lane L -> N=L&15; same K mapping as A (pairs contiguous)
//   C/D (16x16 f32): lane L -> N=L&15; VGPR i -> M = i + 8*hg
// ---------------------------------------------------------------------------
__global__ __launch_bounds__(256)
void pn_transform_gemm(const float* __restrict__ X,
                       const float* __restrict__ W,
                       const float* __restrict__ bias,
                       float* __restrict__ Y) {
    const int K = DK;
    const int t    = threadIdx.x;
    const int wave = t >> 5;
    const int lane = t & 31;
    const int tile = blockIdx.x * 8 + wave;   // 2048 tiles = 128 Mtiles x 16 Ntiles
    const int Mt = tile >> 4;
    const int Nt = tile & 15;

    const int mrow = Mt * 16 + (lane & 15);
    const int ncol = Nt * 16 + (lane & 15);
    const int hg   = lane >> 4;               // half-group (lanes 0-15 vs 16-31)

    v8f acc = {0.f, 0.f, 0.f, 0.f, 0.f, 0.f, 0.f, 0.f};

    const float* __restrict__ xrow = X + (size_t)mrow * K;
    const float* __restrict__ wcol = W + (size_t)ncol * K;

#pragma unroll
    for (int kk = 0; kk < K; kk += 32) {
        v16bf a, b;
#pragma unroll
        for (int j = 0; j < 8; ++j) {
            const int kbA = (j < 4) ? (2 * j + 8 * hg)
                                    : (16 + 2 * (j - 4) + 8 * hg);
            const float2 fa = *(const float2*)(xrow + kk + kbA);
            a[2 * j]     = (__bf16)fa.x;
            a[2 * j + 1] = (__bf16)fa.y;

            const int kbB = kbA;
            const float2 fb = *(const float2*)(wcol + kk + kbB);
            b[2 * j]     = (__bf16)fb.x;
            b[2 * j + 1] = (__bf16)fb.y;
        }
        acc = __builtin_amdgcn_wmma_f32_16x16x32_bf16(
            /*neg_a=*/false, a, /*neg_b=*/false, b,
            /*c_mod=*/(short)0, acc, /*reuse_a=*/false, /*reuse_b=*/false);
    }

    const float bv    = bias[ncol];
    const int   mbase = Mt * 16 + 8 * hg;
#pragma unroll
    for (int i = 0; i < 8; ++i)
        Y[(size_t)(mbase + i) * HH + ncol] = acc[i] + bv;
}

// ---------------------------------------------------------------------------
// Fused xi + masked log-softmax. One 256-thread block per (k, n).
// xi[v] = sum_h tanh(kt[k,n,h] + vt[v,n,h]) * vw[h] + vb
// out[k,n,v] = xi[v] - log(sum_v exp(xi[v])*mask[v,n]), log(1)=0 if sum==0.
// Never materializes the (Lk,Lv,N,H) tensor (256 MB in the reference dataflow).
// Lane L owns h in [8L, 8L+8): two coalesced b128 loads per v-row; kt/vw lane
// slices are loop-invariant and hoisted to registers.
// ---------------------------------------------------------------------------
__global__ __launch_bounds__(256)
void pn_score_kernel(const float* __restrict__ kt,
                     const float* __restrict__ vt,
                     const int*   __restrict__ mask,
                     const float* __restrict__ vw,
                     const float* __restrict__ vb,
                     float* __restrict__ out) {
    __shared__ __align__(16) float kt_s[HH];
    __shared__ __align__(16) float vw_s[HH];
    __shared__ float xi_s[LV];
    __shared__ float red_s[256];

    const int k = blockIdx.x;
    const int n = blockIdx.y;
    const int t = threadIdx.x;
    const int wave = t >> 5;
    const int lane = t & 31;

    kt_s[t] = kt[((size_t)k * NB + n) * HH + t];
    vw_s[t] = vw[t];
    __syncthreads();

    // Per-lane loop-invariant slices (h = 8*lane .. 8*lane+7)
    const float4* kt4 = (const float4*)kt_s;
    const float4* vw4 = (const float4*)vw_s;
    const float4 ka = kt4[2 * lane], kb = kt4[2 * lane + 1];
    const float4 wa = vw4[2 * lane], wb = vw4[2 * lane + 1];

    const float vbias = vb[0];

    // 8 waves, each handles 16 of the 128 v rows; lanes split the H reduction.
    for (int v = wave; v < LV; v += 8) {
        const float* __restrict__ vrow = vt + ((size_t)v * NB + n) * HH;
        if (v + 8 < LV)  // hide next row's latency behind 8 transcendentals
            __builtin_prefetch(vrow + (size_t)8 * NB * HH, 0, 1);

        const float4 va  = ((const float4*)vrow)[2 * lane];
        const float4 vb4 = ((const float4*)vrow)[2 * lane + 1];

        float acc;
        acc  = FAST_TANH(ka.x + va.x)  * wa.x;
        acc += FAST_TANH(ka.y + va.y)  * wa.y;
        acc += FAST_TANH(ka.z + va.z)  * wa.z;
        acc += FAST_TANH(ka.w + va.w)  * wa.w;
        acc += FAST_TANH(kb.x + vb4.x) * wb.x;
        acc += FAST_TANH(kb.y + vb4.y) * wb.y;
        acc += FAST_TANH(kb.z + vb4.z) * wb.z;
        acc += FAST_TANH(kb.w + vb4.w) * wb.w;

        // wave32 shuffle reduction
#pragma unroll
        for (int off = 16; off > 0; off >>= 1)
            acc += __shfl_down(acc, off, 32);
        if (lane == 0) xi_s[v] = acc + vbias;
    }
    __syncthreads();

    // masked exp-sum over v (block tree reduction)
    float contrib = 0.f;
    if (t < LV) contrib = __expf(xi_s[t]) * (float)mask[t * NB + n];
    red_s[t] = contrib;
    __syncthreads();
#pragma unroll
    for (int s = 128; s > 0; s >>= 1) {
        if (t < s) red_s[t] += red_s[t + s];
        __syncthreads();
    }
    const float total = red_s[0];
    const float logz  = (total == 0.f) ? 0.f : __logf(total);

    if (t < LV)
        out[((size_t)k * NB + n) * LV + t] = xi_s[t] - logz;
}

extern "C" void kernel_launch(void* const* d_in, const int* in_sizes, int n_in,
                              void* d_out, int out_size, void* d_ws, size_t ws_size,
                              hipStream_t stream) {
    const float* key   = (const float*)d_in[0];  // (128,16,256)
    const float* value = (const float*)d_in[1];  // (128,16,256)
    const int*   mask  = (const int*)  d_in[2];  // (128,16)
    const float* w1_w  = (const float*)d_in[3];  // (256,256)
    const float* w1_b  = (const float*)d_in[4];  // (256,)
    const float* w2_w  = (const float*)d_in[5];  // (256,256)
    const float* w2_b  = (const float*)d_in[6];  // (256,)
    const float* v_w   = (const float*)d_in[7];  // (1,256)
    const float* v_b   = (const float*)d_in[8];  // (1,)
    float*       out   = (float*)d_out;          // (128,16,128)

    float* kt = (float*)d_ws;                    // 2048*256 f32 = 2 MB
    float* vt = kt + (size_t)LK * NB * HH;       // 2048*256 f32 = 2 MB

    // key_trans / value_trans via bf16 WMMA (f32 accumulate)
    pn_transform_gemm<<<256, 256, 0, stream>>>(key,   w1_w, w1_b, kt);
    pn_transform_gemm<<<256, 256, 0, stream>>>(value, w2_w, w2_b, vt);

    // fused tanh-attention + masked log-softmax
    dim3 grid(LK, NB);
    pn_score_kernel<<<grid, 256, 0, stream>>>(kt, vt, mask, v_w, v_b, out);
}
